// LigerJSDLoss_71159018160660
// MI455X (gfx1250) — matrix operand order repaired and checked by reading, not compiled, not run on hardware
//
#include <hip/hip_runtime.h>
#include <cstdint>

#define N_TOK 2048
#define H_S   2048
#define H_T   4096
#define VOCAB 32000
#define NVT   (VOCAB / 128)          // 250 vocab tiles
#define BETA_ 0.5f
#define IGNORE_INDEX (-100)

typedef __bf16 v8bf  __attribute__((ext_vector_type(8)));
typedef __bf16 v16bf __attribute__((ext_vector_type(16)));
typedef float  v8f   __attribute__((ext_vector_type(8)));
typedef int    v4i   __attribute__((vector_size(16)));

#define AS1 __attribute__((address_space(1)))
#define AS3 __attribute__((address_space(3)))

#if defined(__AMDGCN__) && __has_builtin(__builtin_amdgcn_global_load_async_to_lds_b128)
#define USE_ASYNC 1
#else
#define USE_ASYNC 0
#endif

__device__ __forceinline__ unsigned short f2bf(float f) {
  unsigned int u = __float_as_uint(f);
  u += 0x7FFFu + ((u >> 16) & 1u);          // round-to-nearest-even
  return (unsigned short)(u >> 16);
}

// 16-byte global->LDS copy: async-to-LDS if available, else via registers.
__device__ __forceinline__ void copy16(const unsigned short* __restrict__ g,
                                       unsigned short* l) {
#if USE_ASYNC
  __builtin_amdgcn_global_load_async_to_lds_b128(
      (AS1 v4i*)(uintptr_t)g,
      (AS3 v4i*)(unsigned)(uintptr_t)l, 0, 0);
#else
  *(uint4*)l = *(const uint4*)g;
#endif
}

__device__ __forceinline__ void wait_async() {
#if USE_ASYNC
#if __has_builtin(__builtin_amdgcn_s_wait_asynccnt)
  __builtin_amdgcn_s_wait_asynccnt(0);
#else
  asm volatile("s_wait_asynccnt 0" ::: "memory");
#endif
#endif
}

__device__ __forceinline__ v16bf lds_load_v16(const unsigned short* p) {
  v8bf lo = *(const v8bf*)p;
  v8bf hi = *(const v8bf*)(p + 8);
  return __builtin_shufflevector(lo, hi, 0,1,2,3,4,5,6,7,8,9,10,11,12,13,14,15);
}

// ===========================================================================
// Pack kernels: fp32 -> bf16 in WMMA operand order (run once, memory bound).
// A layout (16-row tiles): off = rowTile*16*K + (chunk*32 + lane)*16 + pos
// B layout (128-col x 128-K stages of 16384 ushorts):
//   off = (colTile*(K/128)+kStage)*16384 + ((colGroup*4+chunk)*32+lane)*16+pos
// ===========================================================================
__global__ void __launch_bounds__(256)
pack_A(const float* __restrict__ X, unsigned short* __restrict__ Xp, int R, int K) {
  const int gpr = K >> 3;
  const int total = R * gpr;
  for (int g = blockIdx.x * blockDim.x + threadIdx.x; g < total;
       g += gridDim.x * blockDim.x) {
    int row = g / gpr;
    int kk  = (g % gpr) << 3;
    const float4* src = (const float4*)(X + (size_t)row * K + kk);
    float4 f0 = src[0], f1 = src[1];
    unsigned short h[8] = {f2bf(f0.x), f2bf(f0.y), f2bf(f0.z), f2bf(f0.w),
                           f2bf(f1.x), f2bf(f1.y), f2bf(f1.z), f2bf(f1.w)};
    int rt = row >> 4, r = row & 15;
    int chunk = kk >> 5, kp0 = kk & 31;
    int lane = r + 16 * ((kp0 >> 3) & 1);
    int pos0 = 8 * (kp0 >> 4);
    size_t off = (size_t)rt * 16 * K + (size_t)(chunk * 32 + lane) * 16 + pos0;
    *(uint4*)(Xp + off) = *(uint4*)h;
  }
}

__global__ void __launch_bounds__(256)
pack_B(const float* __restrict__ W, unsigned short* __restrict__ Wp, int V, int K) {
  const int gpr = K >> 3;
  const int total = V * gpr;
  for (int g = blockIdx.x * blockDim.x + threadIdx.x; g < total;
       g += gridDim.x * blockDim.x) {
    int v  = g / gpr;
    int kk = (g % gpr) << 3;
    const float4* src = (const float4*)(W + (size_t)v * K + kk);
    float4 f0 = src[0], f1 = src[1];
    unsigned short h[8] = {f2bf(f0.x), f2bf(f0.y), f2bf(f0.z), f2bf(f0.w),
                           f2bf(f1.x), f2bf(f1.y), f2bf(f1.z), f2bf(f1.w)};
    int colTile = v >> 7, col = v & 127, colGroup = col >> 4, c = col & 15;
    int kStage = kk >> 7, kl = kk & 127, chunk = kl >> 5, kp0 = kl & 31;
    int lane = c + 16 * ((kp0 >> 3) & 1);
    int pos0 = 8 * (kp0 >> 4);                 // 0 or 8
    size_t off = ((size_t)(colTile * (K >> 7) + kStage)) * 16384
               + (size_t)((colGroup * 4 + chunk) * 32 + lane) * 16 + pos0;
    *(uint4*)(Wp + off) = *(uint4*)h;
  }
}

// ===========================================================================
// GEMM plumbing: stage 32KB operand-ready W tiles, double-buffered async.
// ===========================================================================
__device__ __forceinline__ void stage_tile(const unsigned short* __restrict__ g,
                                           unsigned short* l) {
#pragma unroll
  for (int j = 0; j < 8; ++j) {
    int idx = (threadIdx.x + j * 256) << 3;    // 16384 ushorts total
    copy16(g + idx, l + idx);
  }
}

__device__ __forceinline__ void load_A_packed(const unsigned short* __restrict__ Xp,
                                              int rowTile, int K,
                                              unsigned short* ldsA) {
  const unsigned short* src = Xp + (size_t)rowTile * 16 * K;
  const int total16 = (16 * K) >> 3;
  for (int i = threadIdx.x; i < total16; i += blockDim.x)
    copy16(src + (i << 3), ldsA + (i << 3));
}

// One vocab tile (128 cols) x full K, double-buffered over 128-wide K stages.
__device__ __forceinline__ v8f gemm_tile_K(const unsigned short* __restrict__ ldsA,
                                           const unsigned short* __restrict__ tileBase,
                                           unsigned short* ldsB0, unsigned short* ldsB1,
                                           int nStages, int wave, int lane) {
  v8f acc = {0.f, 0.f, 0.f, 0.f, 0.f, 0.f, 0.f, 0.f};
  stage_tile(tileBase, ldsB0);
  wait_async();
  __syncthreads();
  for (int s = 0; s < nStages; ++s) {
    unsigned short* cur = (s & 1) ? ldsB1 : ldsB0;
    unsigned short* nxt = (s & 1) ? ldsB0 : ldsB1;
    if (s + 1 < nStages)
      stage_tile(tileBase + (size_t)(s + 1) * 16384, nxt);   // overlaps WMMAs
    v16bf a[4], b[4];
#pragma unroll
    for (int c = 0; c < 4; ++c) {               // hoist all LDS operand loads
      a[c] = lds_load_v16(ldsA + ((s * 4 + c) * 32 + lane) * 16);
      b[c] = lds_load_v16(cur + ((wave * 4 + c) * 32 + lane) * 16);
    }
#pragma unroll
    for (int c = 0; c < 4; ++c)
      acc = __builtin_amdgcn_wmma_f32_16x16x32_bf16(false, a[c], false, b[c],
                                                    (short)0, acc, false, false);
    wait_async();
    __syncthreads();
  }
  return acc;
}

// ===========================================================================
// Pass 1: fused GEMM + online logsumexp -> per-row logZ; store f16 logits.
// Logit layout: ((rowTile*NVT + vTile)*8 + wave)*256 + r*32 + lane
// ===========================================================================
__global__ void __launch_bounds__(256)
fused_logit_stats(const unsigned short* __restrict__ Xp,
                  const unsigned short* __restrict__ Wp,
                  int K, float* __restrict__ logZ,
                  unsigned short* __restrict__ Lout) {
  extern __shared__ char smem[];
  unsigned short* ldsA  = (unsigned short*)smem;         // 16*K
  unsigned short* ldsB0 = ldsA + 16 * K;                 // 16384
  unsigned short* ldsB1 = ldsB0 + 16384;                 // 16384
  float* red_m = (float*)(ldsB1 + 16384);                // [8][16]
  float* red_s = red_m + 8 * 16;

  const int tid = threadIdx.x, wave = tid >> 5, lane = tid & 31;
  const int nStages = K >> 7;

  load_A_packed(Xp, blockIdx.x, K, ldsA);

  float m[8], s[8];
#pragma unroll
  for (int r = 0; r < 8; ++r) { m[r] = -3.0e38f; s[r] = 0.f; }

  for (int vt = 0; vt < NVT; ++vt) {
    const unsigned short* tileBase = Wp + (size_t)vt * nStages * 16384;
    v8f acc = gemm_tile_K(ldsA, tileBase, ldsB0, ldsB1, nStages, wave, lane);
    size_t base = (((size_t)blockIdx.x * NVT + vt) * 8 + wave) * 256 + lane;
#pragma unroll
    for (int r = 0; r < 8; ++r) {
      float x = acc[r];
      union { _Float16 h; unsigned short u; } cv;
      cv.h = (_Float16)x;
      Lout[base + r * 32] = cv.u;              // coalesced b16 stores
      float nm = fmaxf(m[r], x);               // online logsumexp
      s[r] = s[r] * __expf(m[r] - nm) + __expf(x - nm);
      m[r] = nm;
    }
  }
#pragma unroll
  for (int off = 1; off < 16; off <<= 1) {     // 16 lanes hold the same row
#pragma unroll
    for (int r = 0; r < 8; ++r) {
      float om = __shfl_xor(m[r], off, 32);
      float os = __shfl_xor(s[r], off, 32);
      float nm = fmaxf(m[r], om);
      s[r] = s[r] * __expf(m[r] - nm) + os * __expf(om - nm);
      m[r] = nm;
    }
  }
  if ((lane & 15) == 0) {
    int rh = (lane >> 4) * 8;
#pragma unroll
    for (int r = 0; r < 8; ++r) {
      red_m[wave * 16 + rh + r] = m[r];
      red_s[wave * 16 + rh + r] = s[r];
    }
  }
  __syncthreads();
  if (tid < 16) {
    float fm = -3.0e38f, fs = 0.f;
#pragma unroll
    for (int w = 0; w < 8; ++w) {
      float om = red_m[w * 16 + tid];
      float os = red_s[w * 16 + tid];
      float nm = fmaxf(fm, om);
      fs = fs * __expf(fm - nm) + os * __expf(om - nm);
      fm = nm;
    }
    logZ[blockIdx.x * 16 + tid] = fm + __logf(fs);
  }
}

// ===========================================================================
// Pass 2: streaming elementwise JSD over stored logits (no GEMM).
// Each 8-element group shares one row -> one logZ/mask lookup, b128 loads.
// ===========================================================================
__global__ void __launch_bounds__(256)
elementwise_jsd(const unsigned short* __restrict__ Ls,
                const unsigned short* __restrict__ Lt,
                const float* __restrict__ logZs, const float* __restrict__ logZt,
                const int* __restrict__ target, float* __restrict__ acc_out) {
  const int groups = (N_TOK / 16) * NVT * 8 * 32;      // E/8 = 8,192,000
  const int groupsPerRowTile = NVT * 8 * 32;           // 64,000
  float jacc = 0.f;
  for (int g = blockIdx.x * blockDim.x + threadIdx.x; g < groups;
       g += gridDim.x * blockDim.x) {
    size_t base = (size_t)g << 3;
    int lane0 = (int)(base & 31);
    int rr    = (int)((base >> 5) & 7);
    int rowTile = g / groupsPerRowTile;
    int row = rowTile * 16 + rr + 8 * (lane0 >> 4);
    float lzs = logZs[row], lzt = logZt[row];
    float mk  = (target[row] != IGNORE_INDEX) ? 1.0f : 0.0f;
    uint4 us = *(const uint4*)(Ls + base);
    uint4 ut = *(const uint4*)(Lt + base);
    const unsigned short* hs = (const unsigned short*)&us;
    const unsigned short* ht = (const unsigned short*)&ut;
#pragma unroll
    for (int e = 0; e < 8; ++e) {
      union { unsigned short u; _Float16 h; } cs, ct;
      cs.u = hs[e]; ct.u = ht[e];
      float log_q = (float)cs.h - lzs;
      float log_p = (float)ct.h - lzt;
      float q  = __expf(log_q);
      float p  = __expf(log_p);
      float mm = BETA_ * p + (1.f - BETA_) * q;
      if (mm > 0.f) {
        float log_m = __logf(mm);
        jacc += mk * (BETA_ * p * (log_p - log_m) +
                      (1.f - BETA_) * q * (log_q - log_m));
      }
    }
  }
  __shared__ float wred[8];
  const int tid = threadIdx.x, wave = tid >> 5, lane = tid & 31;
#pragma unroll
  for (int off = 1; off < 32; off <<= 1) jacc += __shfl_xor(jacc, off, 32);
  if (lane == 0) wred[wave] = jacc;
  __syncthreads();
  if (tid == 0) {
    float tot = 0.f;
#pragma unroll
    for (int w = 0; w < 8; ++w) tot += wred[w];
    atomicAdd(acc_out, tot);
  }
}

__global__ void init_acc(float* __restrict__ acc) { acc[0] = 0.f; }

__global__ void __launch_bounds__(256)
finalize_loss(const float* __restrict__ acc, const int* __restrict__ target,
              float* __restrict__ out) {
  __shared__ int cnt_red[8];
  int tid = threadIdx.x;
  int cnt = 0;
  for (int i = tid; i < N_TOK; i += 256) cnt += (target[i] != IGNORE_INDEX) ? 1 : 0;
#pragma unroll
  for (int off = 1; off < 32; off <<= 1) cnt += __shfl_xor(cnt, off, 32);
  if ((tid & 31) == 0) cnt_red[tid >> 5] = cnt;
  __syncthreads();
  if (tid == 0) {
    int t = 0;
#pragma unroll
    for (int w = 0; w < 8; ++w) t += cnt_red[w];
    if (t < 1) t = 1;
    out[0] = acc[0] / (float)t;
  }
}

extern "C" void kernel_launch(void* const* d_in, const int* in_sizes, int n_in,
                              void* d_out, int out_size, void* d_ws, size_t ws_size,
                              hipStream_t stream) {
  const float* student = (const float*)d_in[0];   // [N, H/2]
  const float* W_s     = (const float*)d_in[1];   // [V, H/2]
  const float* teacher = (const float*)d_in[2];   // [N, H]
  const float* W_t     = (const float*)d_in[3];   // [V, H]
  const int*   target  = (const int*)d_in[4];     // [N]

  float* fws   = (float*)d_ws;
  float* acc   = fws;                  // [1]
  float* logZs = fws + 16;             // [2048]
  float* logZt = fws + 16 + 2048;      // [2048]
  unsigned short* XpS = (unsigned short*)(fws + 16 + 4096);     // 16B aligned
  unsigned short* XpT = XpS + (size_t)N_TOK * H_S;
  unsigned short* WpS = XpT + (size_t)N_TOK * H_T;
  unsigned short* WpT = WpS + (size_t)VOCAB * H_S;
  unsigned short* Ls  = WpT + (size_t)VOCAB * H_T;              // [N*V] f16
  unsigned short* Lt  = Ls + (size_t)N_TOK * VOCAB;             // [N*V] f16

  const size_t smem_stats_s = (size_t)16 * H_S * 2 + 2 * 32768 + 2 * 8 * 16 * 4;
  const size_t smem_stats_t = (size_t)16 * H_T * 2 + 2 * 32768 + 2 * 8 * 16 * 4;

  (void)hipFuncSetAttribute((const void*)fused_logit_stats,
                            hipFuncAttributeMaxDynamicSharedMemorySize,
                            (int)smem_stats_t);

  pack_A<<<2048, 256, 0, stream>>>(student, XpS, N_TOK, H_S);
  pack_A<<<4096, 256, 0, stream>>>(teacher, XpT, N_TOK, H_T);
  pack_B<<<8192, 256, 0, stream>>>(W_s, WpS, VOCAB, H_S);
  pack_B<<<8192, 256, 0, stream>>>(W_t, WpT, VOCAB, H_T);
  init_acc<<<1, 1, 0, stream>>>(acc);

  fused_logit_stats<<<N_TOK / 16, 256, smem_stats_s, stream>>>(XpS, WpS, H_S, logZs, Ls);
  fused_logit_stats<<<N_TOK / 16, 256, smem_stats_t, stream>>>(XpT, WpT, H_T, logZt, Lt);
  elementwise_jsd<<<4096, 256, 0, stream>>>(Ls, Lt, logZs, logZt, target, acc);
  finalize_loss<<<1, 256, 0, stream>>>(acc, target, (float*)d_out);
}
